// MultiHeadAttention_37151467110607
// MI455X (gfx1250) — compile-verified, matrix operand order checked by reading, not run on hardware
//
#include <hip/hip_runtime.h>
#include <hip/hip_bf16.h>
#include <stdint.h>

// ---------------------------------------------------------------------------
// MHA block: B=4, S=1024, D_MODEL=1024, H=16, D_K=D_V=64
// out tuple: (ln_out [B,S,1024] f32, attn [B,H,S,S] f32) concatenated in d_out
// ---------------------------------------------------------------------------

typedef _Float16 half16 __attribute__((ext_vector_type(16)));
typedef float    float8 __attribute__((ext_vector_type(8)));
typedef int      v4i_t  __attribute__((vector_size(16)));

#define LDSK 40  // padded k-stride in halves (32+8) -> 80B row, 16B aligned

// ---- CDNA5 async global->LDS copy (ASYNCcnt) ------------------------------
__device__ __forceinline__ void async_copy_b128(const void* gsrc, void* ldst) {
#if __has_builtin(__builtin_amdgcn_global_load_async_to_lds_b128)
  __builtin_amdgcn_global_load_async_to_lds_b128(
      (__attribute__((address_space(1))) v4i_t*)gsrc,
      (__attribute__((address_space(3))) v4i_t*)ldst,
      0, 0);
#else
  const uint64_t ga = (uint64_t)(uintptr_t)gsrc;
  const uint32_t la = (uint32_t)(uintptr_t)ldst;  // low 32 bits = LDS offset
  asm volatile("global_load_async_to_lds_b128 %0, %1, off"
               :: "v"(la), "v"(ga) : "memory");
#endif
}

__device__ __forceinline__ void wait_asynccnt0() {
#if __has_builtin(__builtin_amdgcn_s_wait_asynccnt)
  __builtin_amdgcn_s_wait_asynccnt(0);
#else
  asm volatile("s_wait_asynccnt 0" ::: "memory");
#endif
}

// Load one WMMA 16-bit A/B fragment for a lane whose "row" (A: m-row, B: n-col)
// data is contiguous over k (32 halves).  CDNA5 layout (ISA 7.12.2):
//   lanes 0-15 : k = {0..7} in v0..3 halves, {16..23} in v4..7 halves
//   lanes 16-31: k = {8..15},            {24..31}
__device__ __forceinline__ half16 frag_load_row(const _Float16* __restrict__ row, int hi) {
  half16 f;
  const int base = hi ? 8 : 0;
#pragma unroll
  for (int j = 0; j < 8; ++j) f[j] = row[base + j];
#pragma unroll
  for (int j = 0; j < 8; ++j) f[8 + j] = row[16 + base + j];
  return f;
}

// ---------------------------------------------------------------------------
// One-time weight prep: convert f32 [K][N] -> f16 transposed [N][K]
// so GEMM B tiles are k-contiguous per output column (async-copyable).
// ---------------------------------------------------------------------------
__global__ void convert_transpose_f16(const float* __restrict__ in,
                                      _Float16* __restrict__ out, int Kd, int Nd) {
  int i = blockIdx.x * blockDim.x + threadIdx.x;
  if (i < Kd * Nd) {
    const int k = i / Nd, n = i % Nd;
    out[(long long)n * Kd + k] = (_Float16)in[i];
  }
}

// ---------------------------------------------------------------------------
// Generic batched WMMA GEMM:  C[M,N] = A[M,K] @ Bt[N,K]^T
//   A : f32 (converted while staging) or f16 (async-copied to LDS)
//   Bt: f16, k-major (row n holds K contiguous halves, stride ldb);
//       async-copied to LDS with no transpose.
//   OUT_MODE: 0 = f32 row-major, 1 = f16 row-major,
//             2 = f16 per-head transposed (Vt[b][h][d][s], S=1024, DH=64)
// Block tile 128 x BN, k-step 32, 256 threads = 8 waves; wave w owns the
// 16-row strip at w*16 with BN/16 wmma tiles across N.
// Batch offset for pointer P: (z/innerCnt)*pOuter + (z%innerCnt)*pInner.
// All dims multiples of tile sizes for every use here (no bounds checks).
// ---------------------------------------------------------------------------
template <bool A_IS_F32, int OUT_MODE, int BN>
__global__ __launch_bounds__(256) void gemm_wmma(
    const void* __restrict__ Av, const _Float16* __restrict__ Bt,
    void* __restrict__ Cv, int K, int lda, int ldb, int ldc,
    long long aOuter, long long aInner, long long bOuter, long long bInner,
    long long cOuter, long long cInner, int innerCnt) {
  constexpr int BM = 128;
  constexpr int BK = 32;
  constexpr int NT = BN / 16;        // wmma tiles per wave across N
  constexpr int BCH = BN * 4 / 256;  // 16B B-chunks per thread (2 or 1)

  __shared__ _Float16 lsA[BM * LDSK];
  __shared__ _Float16 lsB[BN * LDSK];

  const int z = blockIdx.z;
  const long long offA = (long long)(z / innerCnt) * aOuter + (long long)(z % innerCnt) * aInner;
  const long long offB = (long long)(z / innerCnt) * bOuter + (long long)(z % innerCnt) * bInner;
  const long long offC = (long long)(z / innerCnt) * cOuter + (long long)(z % innerCnt) * cInner;

  const int tid  = threadIdx.x;
  const int wave = tid >> 5;
  const int lane = tid & 31;
  const int hi   = lane >> 4;
  const int ln16 = lane & 15;
  const int rowBase = wave * 16;

  const long long blockM = (long long)blockIdx.x * BM;
  const long long blockN = (long long)blockIdx.y * BN;

  const float*    Af = (const float*)Av + offA;
  const _Float16* Ah = (const _Float16*)Av + offA;
  const _Float16* Bp = Bt + offB;

  // cooperative-load coordinates
  const int ar = tid >> 1;        // A tile row 0..127
  const int ak = (tid & 1) * 16;  // A tile k half 0/16

  float8 acc[NT];
  const float8 zero = {};
#pragma unroll
  for (int j = 0; j < NT; ++j) acc[j] = zero;

  for (int k0 = 0; k0 < K; k0 += BK) {
    // ---- stage A tile --------------------------------------------------
    if constexpr (A_IS_F32) {
      const float* src = Af + (blockM + ar) * (long long)lda + k0 + ak;
#pragma unroll
      for (int i = 0; i < 16; ++i) lsA[ar * LDSK + ak + i] = (_Float16)src[i];
    } else {
      const _Float16* src = Ah + (blockM + ar) * (long long)lda + k0 + ak;
#pragma unroll
      for (int c = 0; c < 2; ++c)  // 2 x 16B per thread
        async_copy_b128(src + c * 8, &lsA[ar * LDSK + ak + c * 8]);
    }
    // ---- stage B tile (k-major, no transpose): async b128 chunks -------
#pragma unroll
    for (int i = 0; i < BCH; ++i) {
      const int c  = tid + i * 256;   // chunk id 0 .. BN*4-1
      const int n  = c >> 2;          // tile n-row
      const int kc = c & 3;           // 8-half chunk within row
      async_copy_b128(Bp + (blockN + n) * (long long)ldb + k0 + kc * 8,
                      &lsB[n * LDSK + kc * 8]);
    }
    wait_asynccnt0();
    __syncthreads();

    // ---- wmma over the staged tiles ------------------------------------
    const half16 a = frag_load_row(&lsA[(rowBase + ln16) * LDSK], hi);
#pragma unroll
    for (int j = 0; j < NT; ++j) {
      const half16 b = frag_load_row(&lsB[(j * 16 + ln16) * LDSK], hi);
      acc[j] = __builtin_amdgcn_wmma_f32_16x16x32_f16(false, a, false, b,
                                                      (short)0, acc[j], false, false);
    }
    __syncthreads();
  }

  // ---- epilogue: C layout n = lane%16, vgpr v -> m = v + 8*(lane/16) ---
#pragma unroll
  for (int j = 0; j < NT; ++j) {
    const long long col = blockN + j * 16 + ln16;
#pragma unroll
    for (int v = 0; v < 8; ++v) {
      const long long row = blockM + rowBase + v + 8 * hi;
      if constexpr (OUT_MODE == 0) {
        ((float*)Cv + offC)[row * ldc + col] = acc[j][v];
      } else if constexpr (OUT_MODE == 1) {
        ((_Float16*)Cv + offC)[row * ldc + col] = (_Float16)acc[j][v];
      } else {
        // Vt[b][h][d][s]: row = b*1024+s (token), col = h*64+d
        const long long bidx = row >> 10, sidx = row & 1023;
        const long long hidx = col >> 6,  didx = col & 63;
        ((_Float16*)Cv + offC)[(((bidx * 16 + hidx) * 64 + didx) << 10) + sidx] =
            (_Float16)acc[j][v];
      }
    }
  }
}

// ---------------------------------------------------------------------------
// Fused scores + res_att bias + mask + softmax.
// One workgroup per (b, h, 16-query strip).  Wave w owns key columns
// [w*128, w*128+128): 8 wmma tiles in registers; softmax reduction is shfl
// within 16-lane halves + LDS across the 8 waves.  Writes fp32 attn tuple out.
// ---------------------------------------------------------------------------
__global__ __launch_bounds__(256) void attn_softmax_kernel(
    const _Float16* __restrict__ Qh, const _Float16* __restrict__ Kh,
    const float* __restrict__ resatt, const int* __restrict__ mask,
    float* __restrict__ attnOut) {
  constexpr int S = 1024, H = 16, DH = 64, DM = 1024;
  const int q0 = blockIdx.x * 16;
  const int h  = blockIdx.y;
  const int b  = blockIdx.z;
  const int tid  = threadIdx.x;
  const int wave = tid >> 5;
  const int lane = tid & 31;
  const int hi   = lane >> 4;
  const int ln16 = lane & 15;

  __shared__ float redMax[8][16];
  __shared__ float redSum[8][16];

  const int colBase = wave * 128;

  // prefetch the res_att rows this lane will consume (global_prefetch_b8)
#pragma unroll
  for (int v = 0; v < 8; ++v)
    __builtin_prefetch(resatt + ((long long)(b * H + h) * S + q0 + v + 8 * hi) * S + colBase, 0, 1);

  // A fragments: Q rows q0..q0+15, d = 0..63 (two k-steps of 32)
  const _Float16* qrow = Qh + ((long long)(b * S + q0 + ln16)) * DM + h * DH;
  const half16 a0 = frag_load_row(qrow, hi);
  const half16 a1 = frag_load_row(qrow + 32, hi);

  const float8 zero = {};
  float sc[8][8];

#pragma unroll
  for (int t = 0; t < 8; ++t) {
    const int n0 = colBase + t * 16;
    const _Float16* krow = Kh + ((long long)(b * S + n0 + ln16)) * DM + h * DH;
    const half16 b0 = frag_load_row(krow, hi);
    const half16 b1 = frag_load_row(krow + 32, hi);
    float8 c = zero;
    c = __builtin_amdgcn_wmma_f32_16x16x32_f16(false, a0, false, b0, (short)0, c, false, false);
    c = __builtin_amdgcn_wmma_f32_16x16x32_f16(false, a1, false, b1, (short)0, c, false, false);
#pragma unroll
    for (int v = 0; v < 8; ++v) {
      const int m = q0 + v + 8 * hi;
      const int n = n0 + ln16;
      float s = c[v] * 0.125f + resatt[((long long)(b * H + h) * S + m) * S + n];
      s = mask[((long long)b * S + m) * S + n] ? -1e9f : s;
      sc[t][v] = s;
    }
  }

  // row max across this wave's 128 cols, then across waves
  float fm[8];
#pragma unroll
  for (int v = 0; v < 8; ++v) {
    float x = sc[0][v];
#pragma unroll
    for (int t = 1; t < 8; ++t) x = fmaxf(x, sc[t][v]);
#pragma unroll
    for (int off = 1; off < 16; off <<= 1) x = fmaxf(x, __shfl_xor(x, off, 32));
    fm[v] = x;  // valid in every lane of the half
  }
  if (ln16 == 0) {
#pragma unroll
    for (int v = 0; v < 8; ++v) redMax[wave][hi * 8 + v] = fm[v];
  }
  __syncthreads();
#pragma unroll
  for (int v = 0; v < 8; ++v) {
    float x = redMax[0][hi * 8 + v];
#pragma unroll
    for (int w = 1; w < 8; ++w) x = fmaxf(x, redMax[w][hi * 8 + v]);
    fm[v] = x;
  }

  // exp + row sum
  float rs[8];
#pragma unroll
  for (int v = 0; v < 8; ++v) {
    float s = 0.f;
#pragma unroll
    for (int t = 0; t < 8; ++t) {
      sc[t][v] = __expf(sc[t][v] - fm[v]);
      s += sc[t][v];
    }
#pragma unroll
    for (int off = 1; off < 16; off <<= 1) s += __shfl_xor(s, off, 32);
    rs[v] = s;
  }
  if (ln16 == 0) {
#pragma unroll
    for (int v = 0; v < 8; ++v) redSum[wave][hi * 8 + v] = rs[v];
  }
  __syncthreads();

#pragma unroll
  for (int v = 0; v < 8; ++v) {
    float s = 0.f;
#pragma unroll
    for (int w = 0; w < 8; ++w) s += redSum[w][hi * 8 + v];
    const float inv = 1.0f / s;
    const int m = q0 + v + 8 * hi;
    float* orow = attnOut + ((long long)(b * H + h) * S + m) * S;
#pragma unroll
    for (int t = 0; t < 8; ++t)
      orow[colBase + t * 16 + ln16] = sc[t][v] * inv;
  }
}

// ---------------------------------------------------------------------------
// out = LayerNorm(fc_out + residual) * gamma + beta ; one block per token row
// ---------------------------------------------------------------------------
__global__ __launch_bounds__(256) void add_layernorm_kernel(
    const float* __restrict__ x, const float* __restrict__ resid,
    const float* __restrict__ gamma, const float* __restrict__ beta,
    float* __restrict__ out) {
  constexpr int D = 1024;
  const long long row = blockIdx.x;
  const int tid = threadIdx.x;
  __shared__ float red[256];

  float v[4];
  float s = 0.f;
#pragma unroll
  for (int i = 0; i < 4; ++i) {
    const int c = tid + i * 256;
    v[i] = x[row * D + c] + resid[row * D + c];
    s += v[i];
  }
  red[tid] = s;
  __syncthreads();
  for (int off = 128; off > 0; off >>= 1) {
    if (tid < off) red[tid] += red[tid + off];
    __syncthreads();
  }
  const float mu = red[0] * (1.0f / D);
  __syncthreads();

  s = 0.f;
#pragma unroll
  for (int i = 0; i < 4; ++i) {
    const float d = v[i] - mu;
    s += d * d;
  }
  red[tid] = s;
  __syncthreads();
  for (int off = 128; off > 0; off >>= 1) {
    if (tid < off) red[tid] += red[tid + off];
    __syncthreads();
  }
  const float var = red[0] * (1.0f / D);
  const float r = rsqrtf(var + 1e-5f);
#pragma unroll
  for (int i = 0; i < 4; ++i) {
    const int c = tid + i * 256;
    out[row * D + c] = (v[i] - mu) * r * gamma[c] + beta[c];
  }
}

// ---------------------------------------------------------------------------
extern "C" void kernel_launch(void* const* d_in, const int* in_sizes, int n_in,
                              void* d_out, int out_size, void* d_ws, size_t ws_size,
                              hipStream_t stream) {
  constexpr int B = 4, S = 1024, DM = 1024, H = 16, DH = 64;
  constexpr long long TOK = (long long)B * S;  // 4096 token rows

  const float* inQ = (const float*)d_in[0];
  const float* inK = (const float*)d_in[1];
  const float* inV = (const float*)d_in[2];
  const int*   msk = (const int*)d_in[3];
  const float* res = (const float*)d_in[4];
  const float* Wq  = (const float*)d_in[5];
  const float* Wk  = (const float*)d_in[6];
  const float* Wv  = (const float*)d_in[7];
  const float* Wfc = (const float*)d_in[8];
  const float* gam = (const float*)d_in[9];
  const float* bet = (const float*)d_in[10];

  float* outLN  = (float*)d_out;                       // [B,S,DM]
  float* outAtt = (float*)d_out + TOK * DM;            // [B,H,S,S]

  // workspace layout (56 MiB total)
  char* ws = (char*)d_ws;
  const size_t MB = 1ull << 20;
  _Float16* WqT  = (_Float16*)(ws + 0 * MB);   // 2 MiB each, transposed [N][K]
  _Float16* WkT  = (_Float16*)(ws + 2 * MB);
  _Float16* WvT  = (_Float16*)(ws + 4 * MB);
  _Float16* WfcT = (_Float16*)(ws + 6 * MB);
  _Float16* Qh   = (_Float16*)(ws + 8 * MB);   // 8 MiB, [B,S,H*DH]
  _Float16* Kh   = (_Float16*)(ws + 16 * MB);  // 8 MiB, [B,S,H*DH]
  _Float16* Vt   = (_Float16*)(ws + 24 * MB);  // 8 MiB, [B,H,DH,S]
  _Float16* Ch   = (_Float16*)(ws + 32 * MB);  // 8 MiB, context [B,S,H*DV]
  float*    FC   = (float*)(ws + 40 * MB);     // 16 MiB, fc out f32

  const dim3 blk(256);

  // 1) one-time weight conversion + transpose: f32 [K][N] -> f16 [N][K]
  {
    const int nw = DM * DM;
    const int g = (nw + 255) / 256;
    convert_transpose_f16<<<g, blk, 0, stream>>>(Wq,  WqT,  DM, DM);
    convert_transpose_f16<<<g, blk, 0, stream>>>(Wk,  WkT,  DM, DM);
    convert_transpose_f16<<<g, blk, 0, stream>>>(Wv,  WvT,  DM, DM);
    convert_transpose_f16<<<g, blk, 0, stream>>>(Wfc, WfcT, DM, DM);
  }

  // 2) projections: [4096,1024] = inX(f32) @ W; Q,K -> row-major f16,
  //    V -> per-head transposed f16 (so context-GEMM B is k-major)
  {
    const dim3 grd(TOK / 128, DM / 128, 1);
    gemm_wmma<true, 1, 128><<<grd, blk, 0, stream>>>(
        inQ, WqT, Qh, DM, DM, DM, DM, 0, 0, 0, 0, 0, 0, 1);
    gemm_wmma<true, 1, 128><<<grd, blk, 0, stream>>>(
        inK, WkT, Kh, DM, DM, DM, DM, 0, 0, 0, 0, 0, 0, 1);
    gemm_wmma<true, 2, 128><<<grd, blk, 0, stream>>>(
        inV, WvT, Vt, DM, DM, DM, S, 0, 0, 0, 0, 0, 0, 1);
  }

  // 3) scores + res_att bias + mask + softmax -> attn (fp32 tuple output)
  attn_softmax_kernel<<<dim3(S / 16, H, B), blk, 0, stream>>>(Qh, Kh, res, msk, outAtt);

  // 4) context = attn(f32) @ V : batched over B*H (z/16 = b, z%16 = h)
  //    B operand = Vt slice [DH][S], k(=token)-major, ldb = S
  gemm_wmma<true, 1, 64><<<dim3(S / 128, 1, B * H), blk, 0, stream>>>(
      outAtt, Vt, Ch, S, S, S, DM,
      /*aOuter*/ (long long)H * S * S,  /*aInner*/ (long long)S * S,
      /*bOuter*/ (long long)H * DH * S, /*bInner*/ (long long)DH * S,
      /*cOuter*/ (long long)S * DM,     /*cInner*/ DH, H);

  // 5) fc: [4096,1024] = context(f16, async-staged) @ Wfc, f32 out
  gemm_wmma<false, 0, 128><<<dim3(TOK / 128, DM / 128, 1), blk, 0, stream>>>(
      Ch, WfcT, FC, DM, DM, DM, DM, 0, 0, 0, 0, 0, 0, 1);

  // 6) residual add + layernorm
  add_layernorm_kernel<<<dim3((unsigned)TOK), blk, 0, stream>>>(FC, inQ, gam, bet, outLN);
}
